// ImprovedPixelwiseBoundaryContrastiveLoss_88038239634234
// MI455X (gfx1250) — compile-verified, matrix operand order checked by reading, not run on hardware
//
#include <hip/hip_runtime.h>
#include <hip/hip_bf16.h>

// ---------------- constants mirroring the reference ----------------
#define HDIM   384
#define WDIM   384
#define HW     (HDIM*WDIM)        // 147456
#define CDIM   64
#define PMAXN  2048               // max positives
#define KCAND  20480              // rim candidate pool
#define KNEG   10240              // max negatives
#define RADIUS 5                  // (11-1)/2 dilation radius
#define SHIFT  10.0f              // fixed logsumexp shift: logits = cos_sim/TEMP <= ~10

typedef __attribute__((ext_vector_type(16))) __bf16 v16bf;
typedef __attribute__((ext_vector_type(8)))  float  v8f;

struct Frag32 { uint4 lo, hi; };   // 32 bytes == 16 bf16

static __device__ __forceinline__ int imin(int a, int b){ return a < b ? a : b; }
static __device__ __forceinline__ int imax(int a, int b){ return a > b ? a : b; }

// float -> bf16 (round-to-nearest-even) bit pattern
static __device__ __forceinline__ unsigned short f2bf(float x) {
  unsigned u = __float_as_uint(x);
  unsigned r = u + 0x7FFFu + ((u >> 16) & 1u);
  return (unsigned short)(r >> 16);
}

// float -> order-preserving unsigned key (ascending)
static __device__ __forceinline__ unsigned okey(float x) {
  unsigned u = __float_as_uint(x);
  return (u & 0x80000000u) ? ~u : (u | 0x80000000u);
}

// Load one 16x32 bf16 WMMA operand chunk from a row-major [rows][64] bf16 array.
// Per ISA 16-bit A layout: lanes 0-15 hold row M with K {0..7,16..23} (+32*kchunk),
// lanes 16-31 hold the same rows with K {8..15,24..31}.
// Works for both global and LDS base pointers (address space inferred at inline).
static __device__ __forceinline__ v16bf load_bf16_frag(const unsigned short* base,
                                                       int row, int khalf, int kchunk) {
  const unsigned short* r = base + row * CDIM + kchunk * 32 + khalf * 8;
  Frag32 f;
  f.lo = *reinterpret_cast<const uint4*>(r);       // K + 0..7
  f.hi = *reinterpret_cast<const uint4*>(r + 16);  // K + 16..23
  return __builtin_bit_cast(v16bf, f);
}

// Block-wide (1024 threads, wave32) exclusive scan of a 0/1 flag.
static __device__ unsigned block_scan_flag(int flag, unsigned* swarp, unsigned* s_total, int tid) {
  __syncthreads();
  unsigned lane = tid & 31, wid = tid >> 5;
  unsigned mask = (unsigned)__ballot(flag);
  unsigned lrank = __popc(mask & ((1u << lane) - 1u));
  if (lane == 0) swarp[wid] = __popc(mask);
  __syncthreads();
  if (tid == 0) {
    unsigned a = 0;
    for (int i = 0; i < 32; ++i) { unsigned t = swarp[i]; swarp[i] = a; a += t; }
    *s_total = a;
  }
  __syncthreads();
  return swarp[wid] + lrank;
}

// ---------------- kernel 0: zero per-call accumulators ----------------
__global__ void init_kernel(int* meta, float* loss_sum, int B) {
  int i = threadIdx.x;
  if (i < B) { meta[i * 4 + 3] = 0; loss_sum[i] = 0.f; }
}

// ---------------- kernel 1: fg mask + horizontal dilation ----------------
__global__ void mask_h_kernel(const int* labels, unsigned char* fg, unsigned char* hdil, int total) {
  int i = blockIdx.x * 256 + threadIdx.x;
  if (i >= total) return;
  int x = i % WDIM;
  int f = labels[i] > 0;
  fg[i] = (unsigned char)f;
  int x0 = imax(0, x - RADIUS), x1 = imin(WDIM - 1, x + RADIUS);
  int base = i - x;
  int d = 0;
  for (int xx = x0; xx <= x1; ++xx) d |= (labels[base + xx] > 0);
  hdil[i] = (unsigned char)d;
}

// ---------------- kernel 2: vertical dilation -> rim ----------------
__global__ void mask_v_kernel(const unsigned char* fg, const unsigned char* hdil,
                              unsigned char* rim, int total) {
  int i = blockIdx.x * 256 + threadIdx.x;
  if (i >= total) return;
  int rem = i % HW;
  int y = rem / WDIM;
  int y0 = imax(0, y - RADIUS), y1 = imin(HDIM - 1, y + RADIUS);
  int d = 0;
  for (int yy = y0; yy <= y1; ++yy) d |= hdil[i + (yy - y) * WDIM];
  rim[i] = (unsigned char)(d && !fg[i]);
}

// ---------------- kernel 3: raster-order compaction (one block per image) ----------------
__global__ void compact_kernel(const unsigned char* fg, const unsigned char* rim,
                               int* pidx, int* cidx, int* meta) {
  __shared__ unsigned swarp[32];
  __shared__ unsigned s_total;
  __shared__ unsigned s_base[2];
  int b = blockIdx.x, tid = threadIdx.x;
  if (tid == 0) { s_base[0] = 0; s_base[1] = 0; }
  __syncthreads();
  const unsigned char* fgb = fg + (size_t)b * HW;
  const unsigned char* rimb = rim + (size_t)b * HW;
  for (int chunk = 0; chunk < HW / 1024; ++chunk) {
    int i = chunk * 1024 + tid;
    int f1 = fgb[i];
    unsigned off = block_scan_flag(f1, swarp, &s_total, tid);
    unsigned pos = s_base[0] + off;
    if (f1 && pos < PMAXN) pidx[(size_t)b * PMAXN + pos] = i;
    __syncthreads();
    if (tid == 0) s_base[0] += s_total;
    int f2 = rimb[i];
    off = block_scan_flag(f2, swarp, &s_total, tid);
    pos = s_base[1] + off;
    if (f2 && pos < KCAND) cidx[(size_t)b * KCAND + pos] = i;
    __syncthreads();
    if (tid == 0) s_base[1] += s_total;
    __syncthreads();
  }
  if (tid == 0) {
    int npos  = imin((int)s_base[0], PMAXN);
    int ncand = imin((int)s_base[1], KCAND);
    int nall  = imin(5 * npos, KNEG);      // floor(5.0*npos) exact
    int neff  = imin(nall, ncand);
    meta[b * 4 + 0] = npos;
    meta[b * 4 + 1] = ncand;
    meta[b * 4 + 2] = neff;
  }
}

// ---------------- kernel 4: gather + L2 normalize -> bf16 rows (wave per slot) ----------------
__global__ void gather_norm_kernel(const float* feat, const int* pidx, const int* cidx,
                                   const int* meta, unsigned short* pf, unsigned short* cf, int B) {
  int wid = threadIdx.x >> 5, lane = threadIdx.x & 31;
  int gw = blockIdx.x * 8 + wid;
  const int slots = PMAXN + KCAND;
  int b = gw / slots, s = gw % slots;
  if (b >= B) return;
  int npos = meta[b * 4 + 0], ncand = meta[b * 4 + 1];
  unsigned short* dst; int idx; bool valid;
  if (s < PMAXN) {
    valid = s < npos;
    idx = valid ? pidx[(size_t)b * PMAXN + s] : 0;
    dst = pf + ((size_t)b * PMAXN + s) * CDIM;
  } else {
    int c = s - PMAXN;
    valid = c < ncand;
    idx = valid ? cidx[(size_t)b * KCAND + c] : 0;
    dst = cf + ((size_t)b * KCAND + c) * CDIM;
  }
  float v0 = 0.f, v1 = 0.f;
  if (valid) {
    const float* fb = feat + (size_t)b * CDIM * HW + idx;
    v0 = fb[(size_t)lane * HW];
    v1 = fb[(size_t)(lane + 32) * HW];
  }
  float ss = v0 * v0 + v1 * v1;
#pragma unroll
  for (int off = 1; off < 32; off <<= 1) ss += __shfl_xor(ss, off, 32);
  float inv = valid ? 1.f / fmaxf(sqrtf(ss), 1e-8f) : 0.f;
  dst[lane]      = f2bf(v0 * inv);
  dst[lane + 32] = f2bf(v1 * inv);
}

// ---------------- kernel 5: hard-mining column max via WMMA ----------------
// One block per 128 candidate columns of one image; 8 waves x 16-col tiles.
// A tiles (pf rows) are staged in LDS, shared by all 8 waves.
__global__ void maxsim_kernel(const unsigned short* pf, const unsigned short* cf,
                              const int* meta, float* max_sim) {
  __shared__ __align__(16) unsigned short atile[16 * CDIM];   // 2 KB
  int tid = threadIdx.x;
  int wid = tid >> 5, lane = tid & 31;
  int b = blockIdx.x / (KCAND / 128);
  int colbase = (blockIdx.x % (KCAND / 128)) * 128 + wid * 16;
  int npos = meta[b * 4 + 0], ncand = meta[b * 4 + 1];
  const unsigned short* pfb = pf + (size_t)b * PMAXN * CDIM;
  const unsigned short* cfb = cf + (size_t)b * KCAND * CDIM;
  int khalf = lane >> 4, lr = lane & 15;
  v16bf b0 = load_bf16_frag(cfb, colbase + lr, khalf, 0);
  v16bf b1 = load_bf16_frag(cfb, colbase + lr, khalf, 1);
  float cmax = -__builtin_inff();
  int ptiles = (npos + 15) >> 4;
  for (int pt = 0; pt < ptiles; ++pt) {
    int p = pt * 16;
    __syncthreads();
    if (tid < 128)
      reinterpret_cast<uint4*>(atile)[tid] =
          reinterpret_cast<const uint4*>(pfb + (size_t)p * CDIM)[tid];
    __syncthreads();
    v16bf a0 = load_bf16_frag(atile, lr, khalf, 0);
    v16bf a1 = load_bf16_frag(atile, lr, khalf, 1);
    v8f acc;
#pragma unroll
    for (int i = 0; i < 8; ++i) acc[i] = 0.f;
    acc = __builtin_amdgcn_wmma_f32_16x16x32_bf16(false, a0, false, b0, (short)0, acc, false, false);
    acc = __builtin_amdgcn_wmma_f32_16x16x32_bf16(false, a1, false, b1, (short)0, acc, false, false);
    if (p + 16 <= npos) {
      // fast path: all 16 rows valid -> unconditional max chain
#pragma unroll
      for (int v = 0; v < 8; ++v) cmax = fmaxf(cmax, acc[v]);
    } else {
      // tail tile: mask rows >= npos
#pragma unroll
      for (int v = 0; v < 8; ++v) {
        int grow = p + v + (khalf << 3);               // C/D layout: M = v (+8 for hi half)
        if (grow < npos) cmax = fmaxf(cmax, acc[v]);
      }
    }
  }
  cmax = fmaxf(cmax, __shfl_xor(cmax, 16, 32));        // merge row halves (same column)
  if (lane < 16) {
    int col = colbase + lane;
    max_sim[(size_t)b * KCAND + col] = (col < ncand) ? cmax : -__builtin_inff();
  }
}

// ---------------- kernel 6: radix-select top-neff negatives (one block per image) ----------------
__global__ void select_kernel(const float* max_sim, const int* meta, int* nsel) {
  __shared__ unsigned hist[256];
  __shared__ unsigned swarp[32];
  __shared__ unsigned s_total;
  __shared__ unsigned s_prefix, s_target;
  __shared__ unsigned s_gbase, s_tbase;
  int b = blockIdx.x, tid = threadIdx.x;
  int neff = meta[b * 4 + 2];
  if (neff <= 0) return;
  const float* ms = max_sim + (size_t)b * KCAND;
  if (tid == 0) { s_prefix = 0; s_target = (unsigned)neff; }
  __syncthreads();
  for (int level = 3; level >= 0; --level) {
    if (tid < 256) hist[tid] = 0;
    __syncthreads();
    unsigned pfx = s_prefix;
    for (int j = tid; j < KCAND; j += 1024) {
      unsigned k = okey(ms[j]);
      bool match = (level == 3) || ((k >> ((level + 1) * 8)) == (pfx >> ((level + 1) * 8)));
      if (match) atomicAdd(&hist[(k >> (level * 8)) & 255u], 1u);
    }
    __syncthreads();
    if (tid == 0) {
      unsigned tgt = s_target, cum = 0; int chosen = 0;
      for (int d = 255; d >= 0; --d) {
        unsigned c = hist[d];
        if (cum + c >= tgt) { chosen = d; break; }
        cum += c;
      }
      s_prefix |= (unsigned)chosen << (level * 8);
      s_target = tgt - cum;                           // remaining = ties needed at threshold
    }
    __syncthreads();
  }
  unsigned T = s_prefix;
  unsigned ties = s_target;
  unsigned ngreater = (unsigned)neff - ties;
  if (tid == 0) { s_gbase = 0; s_tbase = 0; }
  __syncthreads();
  for (int chunk = 0; chunk < KCAND / 1024; ++chunk) {
    int j = chunk * 1024 + tid;
    unsigned k = okey(ms[j]);
    int fgt = k > T;
    unsigned off = block_scan_flag(fgt, swarp, &s_total, tid);
    unsigned pos = s_gbase + off;
    if (fgt) nsel[(size_t)b * KNEG + pos] = j;
    __syncthreads();
    if (tid == 0) s_gbase += s_total;
    int ftie = (k == T);
    off = block_scan_flag(ftie, swarp, &s_total, tid);
    pos = s_tbase + off;
    if (ftie && pos < ties) nsel[(size_t)b * KNEG + ngreater + pos] = j;  // lowest-index ties
    __syncthreads();
    if (tid == 0) s_tbase += s_total;
    __syncthreads();
  }
}

// ---------------- kernel 7: gather negative feature rows (wave per row) ----------------
__global__ void gather_neg_kernel(const unsigned short* cf, const int* nsel,
                                  const int* meta, unsigned short* nf) {
  int wid = threadIdx.x >> 5, lane = threadIdx.x & 31;
  int gw = blockIdx.x * 8 + wid;
  int b = gw / KNEG, i = gw % KNEG;
  int neff = meta[b * 4 + 2];
  unsigned* dst = (unsigned*)(nf + ((size_t)b * KNEG + i) * CDIM);
  if (i < neff) {
    int row = nsel[(size_t)b * KNEG + i];
    const unsigned* src = (const unsigned*)(cf + ((size_t)b * KCAND + row) * CDIM);
    dst[lane] = src[lane];
  } else {
    dst[lane] = 0u;
  }
}

// ---------------- kernel 8: contrastive loss via WMMA + fixed-shift softmax ----------------
// One block per 128 anchor rows of one image; 8 waves x 16-row tiles.
// B tiles (pf/nf columns) staged in LDS, shared by all 8 waves.
// Logits are bounded by ~1/TEMP, so exp(logit - SHIFT) is an exact logsumexp shift:
// no running max, no rescaling, cross-lane merge is a plain sum.
__global__ void loss_kernel(const unsigned short* pf, const unsigned short* nf,
                            const int* meta, float* loss_sum, int* meta_mut) {
  __shared__ __align__(16) unsigned short btile[16 * CDIM];   // 2 KB
  int tid = threadIdx.x;
  int wid = tid >> 5, lane = tid & 31;
  int b = blockIdx.x / (PMAXN / 128);
  int rowbase = (blockIdx.x % (PMAXN / 128)) * 128 + wid * 16;
  int npos = meta[b * 4 + 0];
  int neff = meta[b * 4 + 2];
  bool active = rowbase < npos;
  const unsigned short* pfb = pf + (size_t)b * PMAXN * CDIM;
  const unsigned short* nfb = nf + (size_t)b * KNEG * CDIM;
  int khalf = lane >> 4, lr = lane & 15;
  v16bf a0 = load_bf16_frag(pfb, rowbase + lr, khalf, 0);
  v16bf a1 = load_bf16_frag(pfb, rowbase + lr, khalf, 1);
  float s[8], t[8];
#pragma unroll
  for (int v = 0; v < 8; ++v) { s[v] = 0.f; t[v] = 0.f; }
  const float invT = 10.0f;                            // 1 / TEMP
  int myrtile = rowbase >> 4;
  int ppTiles = (npos + 15) >> 4;
  int pnTiles = (neff + 15) >> 4;
  // ---- positive-positive logits ----
  for (int jt = 0; jt < ppTiles; ++jt) {
    int jbase = jt * 16;
    __syncthreads();
    if (tid < 128)
      reinterpret_cast<uint4*>(btile)[tid] =
          reinterpret_cast<const uint4*>(pfb + (size_t)jbase * CDIM)[tid];
    __syncthreads();
    v16bf b0 = load_bf16_frag(btile, lr, khalf, 0);
    v16bf b1 = load_bf16_frag(btile, lr, khalf, 1);
    v8f acc;
#pragma unroll
    for (int i = 0; i < 8; ++i) acc[i] = 0.f;
    acc = __builtin_amdgcn_wmma_f32_16x16x32_bf16(false, a0, false, b0, (short)0, acc, false, false);
    acc = __builtin_amdgcn_wmma_f32_16x16x32_bf16(false, a1, false, b1, (short)0, acc, false, false);
    if (active) {
      if (jt != myrtile && jbase + 16 <= npos) {
        // fast path: no diagonal, all 16 columns valid
#pragma unroll
        for (int v = 0; v < 8; ++v) {
          float logit = acc[v] * invT;
          s[v] += __expf(logit - SHIFT);
          t[v] += logit;
        }
      } else {
        int gcol = jbase + lr;
#pragma unroll
        for (int v = 0; v < 8; ++v) {
          int grow = rowbase + v + (khalf << 3);
          float logit = acc[v] * invT;
          if (gcol < npos && gcol != grow) {
            s[v] += __expf(logit - SHIFT);
            t[v] += logit;
          }
        }
      }
    }
  }
  // ---- positive-negative logits ----
  for (int jt = 0; jt < pnTiles; ++jt) {
    int jbase = jt * 16;
    __syncthreads();
    if (tid < 128)
      reinterpret_cast<uint4*>(btile)[tid] =
          reinterpret_cast<const uint4*>(nfb + (size_t)jbase * CDIM)[tid];
    __syncthreads();
    v16bf b0 = load_bf16_frag(btile, lr, khalf, 0);
    v16bf b1 = load_bf16_frag(btile, lr, khalf, 1);
    v8f acc;
#pragma unroll
    for (int i = 0; i < 8; ++i) acc[i] = 0.f;
    acc = __builtin_amdgcn_wmma_f32_16x16x32_bf16(false, a0, false, b0, (short)0, acc, false, false);
    acc = __builtin_amdgcn_wmma_f32_16x16x32_bf16(false, a1, false, b1, (short)0, acc, false, false);
    if (active) {
      if (jbase + 16 <= neff) {
#pragma unroll
        for (int v = 0; v < 8; ++v) s[v] += __expf(acc[v] * invT - SHIFT);
      } else {
        int gcol = jbase + lr;
        if (gcol < neff) {
#pragma unroll
          for (int v = 0; v < 8; ++v) s[v] += __expf(acc[v] * invT - SHIFT);
        }
      }
    }
  }
  // ---- merge partial sums across the 16 lanes of each half ----
#pragma unroll
  for (int v = 0; v < 8; ++v) {
#pragma unroll
    for (int off = 1; off < 16; off <<= 1) {
      s[v] += __shfl_xor(s[v], off, 16);
      t[v] += __shfl_xor(t[v], off, 16);
    }
  }
  int npairs = npos - 1;
  if (active && npairs > 0) {
#pragma unroll
    for (int v = 0; v < 8; ++v) {
      if (lr == v) {                                    // one writer per row per half
        int grow = rowbase + v + (khalf << 3);
        if (grow < npos && s[v] > 0.f) {
          float denom = SHIFT + __logf(s[v]);
          float li = -(t[v] - (float)npairs * denom) / (float)npairs;
          atomicAdd(&loss_sum[b], li);
          atomicAdd(&meta_mut[b * 4 + 3], 1);
        }
      }
    }
  }
}

// ---------------- kernel 9: finalize scalar ----------------
__global__ void finalize_kernel(const float* loss_sum, const int* meta, float* out, int B) {
  if (threadIdx.x == 0 && blockIdx.x == 0) {
    float total = 0.f; int hs = 0;
    for (int b = 0; b < B; ++b) {
      int c = meta[b * 4 + 3];
      if (c > 0) { total += loss_sum[b] / (float)c; hs++; }
    }
    out[0] = total / (float)(hs > 0 ? hs : 1);
  }
}

extern "C" void kernel_launch(void* const* d_in, const int* in_sizes, int n_in,
                              void* d_out, int out_size, void* d_ws, size_t ws_size,
                              hipStream_t stream) {
  const float* feat  = (const float*)d_in[0];
  const int*   labels = (const int*)d_in[1];
  float* out = (float*)d_out;
  int B = in_sizes[0] / (CDIM * HW);

  char* p = (char*)d_ws;
  auto alloc = [&](size_t bytes) -> char* {
    char* r = p; p += (bytes + 255) & ~(size_t)255; return r;
  };
  unsigned char* fg   = (unsigned char*)alloc((size_t)B * HW);
  unsigned char* hdil = (unsigned char*)alloc((size_t)B * HW);
  unsigned char* rim  = (unsigned char*)alloc((size_t)B * HW);
  int*   pidx     = (int*)alloc((size_t)B * PMAXN * 4);
  int*   cidx     = (int*)alloc((size_t)B * KCAND * 4);
  int*   meta     = (int*)alloc((size_t)B * 4 * 4);
  float* loss_sum = (float*)alloc((size_t)B * 4);
  unsigned short* pf = (unsigned short*)alloc((size_t)B * PMAXN * CDIM * 2);
  unsigned short* cf = (unsigned short*)alloc((size_t)B * KCAND * CDIM * 2);
  unsigned short* nf = (unsigned short*)alloc((size_t)B * KNEG  * CDIM * 2);
  float* max_sim = (float*)alloc((size_t)B * KCAND * 4);
  int*   nsel    = (int*)alloc((size_t)B * KNEG * 4);

  int total = B * HW;
  init_kernel<<<1, 256, 0, stream>>>(meta, loss_sum, B);
  mask_h_kernel<<<(total + 255) / 256, 256, 0, stream>>>(labels, fg, hdil, total);
  mask_v_kernel<<<(total + 255) / 256, 256, 0, stream>>>(fg, hdil, rim, total);
  compact_kernel<<<B, 1024, 0, stream>>>(fg, rim, pidx, cidx, meta);
  gather_norm_kernel<<<B * (PMAXN + KCAND) / 8, 256, 0, stream>>>(feat, pidx, cidx, meta, pf, cf, B);
  maxsim_kernel<<<B * (KCAND / 128), 256, 0, stream>>>(pf, cf, meta, max_sim);
  select_kernel<<<B, 1024, 0, stream>>>(max_sim, meta, nsel);
  gather_neg_kernel<<<B * KNEG / 8, 256, 0, stream>>>(cf, nsel, meta, nf);
  loss_kernel<<<B * (PMAXN / 128), 256, 0, stream>>>(pf, nf, meta, loss_sum, meta);
  finalize_kernel<<<1, 64, 0, stream>>>(loss_sum, meta, out, B);
}
